// MultiHeadAttention_72593537237454
// MI455X (gfx1250) — matrix operand order repaired and checked by reading, not compiled, not run on hardware
//
#include <hip/hip_runtime.h>
#include <hip/hip_bf16.h>

#define HID 1024
#define HEADS 16
#define HEAD_DIM 64
#define NB 4
#define SEQ 4096
#define MTOT (NB * SEQ) /* 16384 rows total */

typedef __attribute__((ext_vector_type(16))) _Float16 v16h;
typedef __attribute__((ext_vector_type(8)))  _Float16 v8h;
typedef __attribute__((ext_vector_type(8)))  float    v8f;
typedef __attribute__((ext_vector_type(4)))  int      v4i;

#ifndef __has_builtin
#define __has_builtin(x) 0
#endif
#if __has_builtin(__builtin_amdgcn_global_load_async_to_lds_b128) && \
    __has_builtin(__builtin_amdgcn_s_wait_asynccnt)
#define HAVE_ASYNC 1
#else
#define HAVE_ASYNC 0
#endif

// ---------------------------------------------------------------------------
// WMMA helpers (gfx1250: D = A(16x32 f16) * B(32x16 f16) + C(16x16 f32))
// ---------------------------------------------------------------------------
__device__ __forceinline__ v8f wmma_f16(v16h a, v16h b, v8f c) {
  // 8 args: (neg_a, A, neg_b, B, c_mod, C, reuse_a, reuse_b)
  return __builtin_amdgcn_wmma_f32_16x16x32_f16(false, a, false, b, (short)0, c,
                                                false, false);
}

// A-fragment: lane<16 holds row m, K = {kb+0..15}
//             lane>=16 holds row m, K = {kb+16..31}
__device__ __forceinline__ v16h load_a_frag(const _Float16* __restrict__ rowp,
                                            int hi) {
  const int lo = hi ? 16 : 0;
  v8h a0 = *(const v8h*)(rowp + lo);
  v8h a1 = *(const v8h*)(rowp + lo + 8);
  v16h a;
#pragma unroll
  for (int i = 0; i < 8; ++i) { a[i] = a0[i]; a[i + 8] = a1[i]; }
  return a;
}

// B-fragment from a pre-transposed weight WT[n][k]: lane holds column n =
// lane&15, contiguous K run of 16 starting at kb + (lane>=16 ? 16 : 0).
__device__ __forceinline__ v16h load_b_frag(const _Float16* __restrict__ base) {
  v8h b0 = *(const v8h*)(base);
  v8h b1 = *(const v8h*)(base + 8);
  v16h b;
#pragma unroll
  for (int i = 0; i < 8; ++i) { b[i] = b0[i]; b[i + 8] = b1[i]; }
  return b;
}

// ---------------------------------------------------------------------------
// 1) fp32 -> f16 conversion for x, and transpose+convert for the 4 weights
// ---------------------------------------------------------------------------
__global__ __launch_bounds__(256) void convert_transpose_kernel(
    const float* __restrict__ x, const float* __restrict__ wq,
    const float* __restrict__ wk, const float* __restrict__ wv,
    const float* __restrict__ wo, _Float16* __restrict__ xh,
    _Float16* __restrict__ wqT, _Float16* __restrict__ wkT,
    _Float16* __restrict__ wvT, _Float16* __restrict__ woT) {
  const size_t idx = (size_t)blockIdx.x * blockDim.x + threadIdx.x;
  if (idx < (size_t)MTOT * HID) xh[idx] = (_Float16)x[idx];
  if (idx < (size_t)HID * HID) {
    const int k = (int)(idx >> 10);
    const int n = (int)(idx & (HID - 1));
    const size_t t = (size_t)n * HID + k;  // WT[n][k] = W[k][n]
    wqT[t] = (_Float16)wq[idx];
    wkT[t] = (_Float16)wk[idx];
    wvT[t] = (_Float16)wv[idx];
    woT[t] = (_Float16)wo[idx];
  }
}

// ---------------------------------------------------------------------------
// 2) Fused Q/K/V projection. Block = 128 threads (4 waves). Each block owns a
//    64(M) x 64(N) tile for ALL THREE weights; the A-fragment is reused across
//    12 WMMAs per K-step.  grid = (MTOT/64, HID/64)
// ---------------------------------------------------------------------------
__global__ __launch_bounds__(128) void qkv_gemm_wmma_kernel(
    const _Float16* __restrict__ xh, const _Float16* __restrict__ wqT,
    const _Float16* __restrict__ wkT, const _Float16* __restrict__ wvT,
    _Float16* __restrict__ qh, _Float16* __restrict__ kh,
    _Float16* __restrict__ vh) {
  const int wave = threadIdx.x >> 5;
  const int lane = threadIdx.x & 31;
  const int lm = lane & 15;
  const int hi = lane >> 4;
  const int m0 = blockIdx.x * 64 + wave * 16;
  const int n0 = blockIdx.y * 64;

  const v8f vzero = {};
  v8f acc[3][4];
#pragma unroll
  for (int w = 0; w < 3; ++w)
#pragma unroll
    for (int nt = 0; nt < 4; ++nt) acc[w][nt] = vzero;

  const _Float16* __restrict__ wT[3] = {wqT, wkT, wvT};
  const _Float16* arow = xh + (size_t)(m0 + lm) * HID;

  for (int kb = 0; kb < HID; kb += 32) {
    const v16h a = load_a_frag(arow + kb, hi);
#pragma unroll
    for (int w = 0; w < 3; ++w) {
#pragma unroll
      for (int nt = 0; nt < 4; ++nt) {
        const _Float16* bbase =
            wT[w] + (size_t)(n0 + nt * 16 + lm) * HID + kb + hi * 16;
        const v16h b = load_b_frag(bbase);
        acc[w][nt] = wmma_f16(a, b, acc[w][nt]);
      }
    }
  }

  _Float16* __restrict__ outp[3] = {qh, kh, vh};
#pragma unroll
  for (int w = 0; w < 3; ++w) {
#pragma unroll
    for (int nt = 0; nt < 4; ++nt) {
      const int col = n0 + nt * 16 + lm;
#pragma unroll
      for (int r = 0; r < 8; ++r) {
        const int row = m0 + r + hi * 8;
        outp[w][(size_t)row * HID + col] = (_Float16)acc[w][nt][r];
      }
    }
  }
}

// ---------------------------------------------------------------------------
// 3) energy partials: part[n][chunk][a][b] = sum over 256 seq rows of
//    q[n,s,a*64+d]*k[n,s,b*64+d].  Fixed-order (deterministic), no atomics.
//    grid = (NB, 16 chunks), block = 256 (thread = (a,b) pair)
// ---------------------------------------------------------------------------
__global__ __launch_bounds__(256) void energy_partial_kernel(
    const _Float16* __restrict__ qh, const _Float16* __restrict__ kh,
    float* __restrict__ part) {
  const int n = blockIdx.x;
  const int chunk = blockIdx.y;
  const int a = threadIdx.x >> 4;
  const int b = threadIdx.x & 15;
  const size_t srow0 = ((size_t)n * SEQ + (size_t)chunk * 256) * HID;
  const _Float16* qbase = qh + srow0 + a * HEAD_DIM;
  const _Float16* kbase = kh + srow0 + b * HEAD_DIM;
  float sum = 0.0f;
  for (int s = 0; s < 256; ++s) {
    const _Float16* qp = qbase + (size_t)s * HID;
    const _Float16* kp = kbase + (size_t)s * HID;
#pragma unroll
    for (int d = 0; d < HEAD_DIM; d += 8) {
      v8h qv = *(const v8h*)(qp + d);
      v8h kv = *(const v8h*)(kp + d);
#pragma unroll
      for (int j = 0; j < 8; ++j) sum += (float)qv[j] * (float)kv[j];
    }
  }
  part[(((size_t)n * 16 + chunk) * 16 + a) * 16 + b] = sum;
}

// ---------------------------------------------------------------------------
// 4) reduce chunks + softmax over b.  1 block, 64 threads (thread = (n,a)).
// ---------------------------------------------------------------------------
__global__ __launch_bounds__(64) void softmax_kernel(
    const float* __restrict__ part, float* __restrict__ attn) {
  const int t = threadIdx.x;
  const int n = t >> 4;
  const int a = t & 15;
  float e[16];
#pragma unroll
  for (int b = 0; b < 16; ++b) {
    float s = 0.0f;
    for (int c = 0; c < 16; ++c)
      s += part[(((size_t)n * 16 + c) * 16 + a) * 16 + b];
    e[b] = s * 0.125f;  // 1/sqrt(HEAD_DIM)
  }
  float mx = e[0];
#pragma unroll
  for (int b = 1; b < 16; ++b) mx = fmaxf(mx, e[b]);
  float sum = 0.0f;
#pragma unroll
  for (int b = 0; b < 16; ++b) {
    e[b] = expf(e[b] - mx);
    sum += e[b];
  }
  const float inv = 1.0f / sum;
#pragma unroll
  for (int b = 0; b < 16; ++b)
    attn[((size_t)n * 16 + a) * 16 + b] = e[b] * inv;
}

// ---------------------------------------------------------------------------
// 5) out[n,a,s,d] = sum_b attn[n,a,b]*v[n,s,b*64+d], stored with the torch
//    reshape fused: row = a*256 + s/16, col = 64*(s%16)+d, as f16 for the
//    final WMMA GEMM.  One block per (n,s) row; v-row staged in LDS via the
//    CDNA5 async global->LDS path when the builtin is available.
// ---------------------------------------------------------------------------
__global__ __launch_bounds__(256) void attn_v_kernel(
    const float* __restrict__ attn, const _Float16* __restrict__ vh,
    _Float16* __restrict__ out2h) {
  __shared__ _Float16 vsh[HID];
  __shared__ float ash[256];
  const int ns = blockIdx.x;  // n*SEQ + s
  const int n = ns >> 12;
  const int s = ns & (SEQ - 1);
  const _Float16* vrow = vh + (size_t)ns * HID;

#if HAVE_ASYNC
  if (threadIdx.x < 128) {  // 128 lanes x 16B = the whole 2KB v-row
    __builtin_amdgcn_global_load_async_to_lds_b128(
        (__attribute__((address_space(1))) v4i*)(vrow + threadIdx.x * 8),
        (__attribute__((address_space(3))) v4i*)(&vsh[threadIdx.x * 8]), 0, 0);
  }
  ash[threadIdx.x] = attn[n * 256 + threadIdx.x];
  __builtin_amdgcn_s_wait_asynccnt(0);
  __syncthreads();
#else
  for (int i = threadIdx.x; i < HID; i += 256) vsh[i] = vrow[i];
  ash[threadIdx.x] = attn[n * 256 + threadIdx.x];
  __syncthreads();
#endif

  const int r_base = s >> 4;
  const int c_base = (s & 15) << 6;
#pragma unroll
  for (int i = 0; i < 4; ++i) {
    const int idx = threadIdx.x + 256 * i;  // 1024 outputs per row
    const int a = idx >> 6;
    const int d = idx & 63;
    float sum = 0.0f;
#pragma unroll
    for (int b = 0; b < 16; ++b)
      sum += ash[a * 16 + b] * (float)vsh[b * 64 + d];
    const int row = a * 256 + r_base;
    const int col = c_base + d;
    out2h[((size_t)n * SEQ + row) * HID + col] = (_Float16)sum;
  }
}

// ---------------------------------------------------------------------------
// 6) final projection: out = out2 @ Wo + bo, fp32 output.
//    grid = (MTOT/64, HID/64), block = 128
// ---------------------------------------------------------------------------
__global__ __launch_bounds__(128) void out_gemm_wmma_kernel(
    const _Float16* __restrict__ ah, const _Float16* __restrict__ woT,
    const float* __restrict__ bo, float* __restrict__ out) {
  const int wave = threadIdx.x >> 5;
  const int lane = threadIdx.x & 31;
  const int lm = lane & 15;
  const int hi = lane >> 4;
  const int m0 = blockIdx.x * 64 + wave * 16;
  const int n0 = blockIdx.y * 64;

  const v8f vzero = {};
  v8f acc[4];
#pragma unroll
  for (int nt = 0; nt < 4; ++nt) acc[nt] = vzero;

  const _Float16* arow = ah + (size_t)(m0 + lm) * HID;
  for (int kb = 0; kb < HID; kb += 32) {
    const v16h a = load_a_frag(arow + kb, hi);
#pragma unroll
    for (int nt = 0; nt < 4; ++nt) {
      const _Float16* bbase =
          woT + (size_t)(n0 + nt * 16 + lm) * HID + kb + hi * 16;
      const v16h b = load_b_frag(bbase);
      acc[nt] = wmma_f16(a, b, acc[nt]);
    }
  }
#pragma unroll
  for (int nt = 0; nt < 4; ++nt) {
    const int col = n0 + nt * 16 + lm;
    const float bias = bo[col];
#pragma unroll
    for (int r = 0; r < 8; ++r) {
      const int row = m0 + r + hi * 8;
      out[(size_t)row * HID + col] = acc[nt][r] + bias;
    }
  }
}

// ---------------------------------------------------------------------------
extern "C" void kernel_launch(void* const* d_in, const int* in_sizes, int n_in,
                              void* d_out, int out_size, void* d_ws,
                              size_t ws_size, hipStream_t stream) {
  (void)in_sizes; (void)n_in; (void)out_size; (void)ws_size;
  const float* x  = (const float*)d_in[0];
  const float* Wq = (const float*)d_in[1];
  const float* Wk = (const float*)d_in[2];
  const float* Wv = (const float*)d_in[3];
  const float* Wo = (const float*)d_in[4];
  const float* bo = (const float*)d_in[5];
  float* out = (float*)d_out;

  char* ws = (char*)d_ws;
  size_t off = 0;
  auto alloc = [&](size_t bytes) -> void* {
    void* p = (void*)(ws + off);
    off = (off + bytes + 255) & ~(size_t)255;
    return p;
  };

  _Float16* xh  = (_Float16*)alloc((size_t)MTOT * HID * 2);
  _Float16* wqT = (_Float16*)alloc((size_t)HID * HID * 2);
  _Float16* wkT = (_Float16*)alloc((size_t)HID * HID * 2);
  _Float16* wvT = (_Float16*)alloc((size_t)HID * HID * 2);
  _Float16* woT = (_Float16*)alloc((size_t)HID * HID * 2);
  _Float16* qh  = (_Float16*)alloc((size_t)MTOT * HID * 2);
  _Float16* kh  = (_Float16*)alloc((size_t)MTOT * HID * 2);
  _Float16* vh  = (_Float16*)alloc((size_t)MTOT * HID * 2);
  _Float16* o2h = (_Float16*)alloc((size_t)MTOT * HID * 2);
  float* part   = (float*)alloc((size_t)NB * 16 * 16 * 16 * sizeof(float));
  float* attn   = (float*)alloc((size_t)NB * 16 * 16 * sizeof(float));

  convert_transpose_kernel<<<(MTOT * HID) / 256, 256, 0, stream>>>(
      x, Wq, Wk, Wv, Wo, xh, wqT, wkT, wvT, woT);
  qkv_gemm_wmma_kernel<<<dim3(MTOT / 64, HID / 64), 128, 0, stream>>>(
      xh, wqT, wkT, wvT, qh, kh, vh);
  energy_partial_kernel<<<dim3(NB, 16), 256, 0, stream>>>(qh, kh, part);
  softmax_kernel<<<1, 64, 0, stream>>>(part, attn);
  attn_v_kernel<<<NB * SEQ, 256, 0, stream>>>(attn, vh, o2h);
  out_gemm_wmma_kernel<<<dim3(MTOT / 64, HID / 64), 128, 0, stream>>>(
      o2h, woT, bo, out);
}
